// EncoderTransformer_35983236006263
// MI455X (gfx1250) — matrix operand order repaired and checked by reading, not compiled, not run on hardware
//
#include <hip/hip_runtime.h>
#include <hip/hip_bf16.h>
#include <math.h>

typedef __bf16 bf16;
typedef __attribute__((ext_vector_type(16))) __bf16 v16bf;
typedef __attribute__((ext_vector_type(8)))  float   v8f;
typedef int v4i_a __attribute__((vector_size(16)));

union Frag { v16bf v; unsigned u[8]; bf16 h[16]; };

// ---- gfx1250 async global->LDS copy (ASYNCcnt) with sync fallback --------
#if defined(__has_builtin)
#if __has_builtin(__builtin_amdgcn_global_load_async_to_lds_b128)
#define HAVE_ASYNC 1
#endif
#if __has_builtin(__builtin_amdgcn_s_wait_asynccnt)
#define HAVE_ASYNCCNT 1
#endif
#endif

__device__ inline void async_copy16(void* lds, const void* g) {
#ifdef HAVE_ASYNC
  __builtin_amdgcn_global_load_async_to_lds_b128(
      (__attribute__((address_space(1))) v4i_a*)g,
      (__attribute__((address_space(3))) v4i_a*)lds, 0, 0);
#else
  *(float4*)lds = *(const float4*)g;
#endif
}

__device__ inline void async_wait0() {
#ifdef HAVE_ASYNC
#ifdef HAVE_ASYNCCNT
  __builtin_amdgcn_s_wait_asynccnt(0);
#else
  asm volatile("s_wait_asynccnt 0x0" ::: "memory");
#endif
#endif
}

// ---- fragment loaders (CDNA5 16x16x32 bf16 layouts, wave32) --------------
// A-matrix 16x32: lane m = lane&15; VGPR i holds K pair at
//   k = (i>=4?16:0) + 8*(lane>=16) + 2*(i&3)
__device__ inline v16bf load_a_frag(const bf16* base, int ldr, int lane) {
  Frag f;
  const bf16* row = base + (lane & 15) * ldr;
  const int hi = (lane >> 4) << 3;
#pragma unroll
  for (int i = 0; i < 8; ++i) {
    int kk = ((i & 4) << 2) + hi + ((i & 3) << 1);
    f.u[i] = *(const unsigned*)(row + kk);
  }
  return f.v;
}
// B-matrix 32x16: lane n = lane&15; same K order as A (k = (i>=4?16:0) + 8*(lane>=16) + 2*(i&3)).
// Source stored transposed: base[n*ldr + k].
__device__ inline v16bf load_b_frag(const bf16* base, int ldr, int lane) {
  Frag f;
  const bf16* row = base + (lane & 15) * ldr + ((lane >> 4) << 3);
#pragma unroll
  for (int i = 0; i < 8; ++i) f.u[i] = *(const unsigned*)(row + ((i & 4) << 2) + ((i & 3) << 1));
  return f.v;
}

__device__ inline v8f wmma_bf16(v16bf a, v16bf b, v8f c) {
  return __builtin_amdgcn_wmma_f32_16x16x32_bf16(false, a, false, b,
                                                 (short)0, c, false, false);
}

// ---- embedding: h = tok_emb[x] + pos_emb, stored bf16 --------------------
__global__ void embed_bf16(const int* __restrict__ x, const float* __restrict__ tok,
                           const float* __restrict__ pos, bf16* __restrict__ hb) {
  size_t gid = (size_t)blockIdx.x * blockDim.x + threadIdx.x;
  int row = (int)(gid >> 10), col = (int)(gid & 1023);
  int tk = x[row];
  float v = tok[(size_t)tk * 1024 + col] + pos[(size_t)(row & 2047) * 1024 + col];
  hb[gid] = (bf16)v;
}

__global__ void cvt_bf16(const float* __restrict__ s, bf16* __restrict__ d, long n) {
  long i = (long)blockIdx.x * blockDim.x + threadIdx.x;
  if (i < n) d[i] = (bf16)s[i];
}

__global__ void zero_loss(float* p) { *p = 0.f; }

// ---- bf16 WMMA GEMM: C[M,N] = A[M,K] x B[K,N] ----------------------------
// 256 thr (8 waves), block tile 128x128, waves 4(M)x2(N), wave tile 32x64.
// mode 0: fp32 logits (NT store) + bias; mode 1: bf16 [b][h][t][d]*scale;
// mode 2: bf16 [b][h][d][t] (V^T).
__global__ void __launch_bounds__(256)
gemm_bf16_wmma(const bf16* __restrict__ A, const bf16* __restrict__ B,
               int K, int ldb, float* __restrict__ outF, bf16* __restrict__ outB,
               const float* __restrict__ bias, int mode, float scale) {
  __shared__ bf16 sA[128][40];
  __shared__ bf16 sB[128][40];   // transposed: sB[n][k]
  const int tid = threadIdx.x, lane = tid & 31, wave = tid >> 5;
  const int wr = wave >> 1, wc = wave & 1;
  const int bm = blockIdx.y * 128, bn = blockIdx.x * 128;

  v8f acc[2][4] = {};
  const int ar = tid >> 1, acs = (tid & 1) * 16;    // A fill: row, col-seg
  const int bkr = tid >> 3, bns = (tid & 7) * 16;   // B fill: k-row, n-seg

  for (int k0 = 0; k0 < K; k0 += 32) {
    if (k0 + 32 < K) {
      __builtin_prefetch(A + (size_t)(bm + ar) * K + k0 + 32 + acs, 0, 1);
      __builtin_prefetch(B + (size_t)(k0 + 32 + bkr) * ldb + bn + bns, 0, 1);
    }
    // A tile: async straight copy (contiguous 16B per lane)
    const bf16* ga = A + (size_t)(bm + ar) * K + k0 + acs;
    async_copy16(&sA[ar][acs], ga);
    async_copy16(&sA[ar][acs + 8], ga + 8);
    // B tile: transpose scatter into sB[n][k]
    const bf16* gb = B + (size_t)(k0 + bkr) * ldb + bn + bns;
    float4 bv0 = *(const float4*)gb;
    float4 bv1 = *(const float4*)(gb + 8);
    const bf16* bp0 = (const bf16*)&bv0;
    const bf16* bp1 = (const bf16*)&bv1;
#pragma unroll
    for (int j = 0; j < 8; ++j) {
      sB[bns + j][bkr]     = bp0[j];
      sB[bns + 8 + j][bkr] = bp1[j];
    }
    async_wait0();
    __syncthreads();

    v16bf a0 = load_a_frag(&sA[wr * 32][0],      40, lane);
    v16bf a1 = load_a_frag(&sA[wr * 32 + 16][0], 40, lane);
#pragma unroll
    for (int tj = 0; tj < 4; ++tj) {
      v16bf b = load_b_frag(&sB[wc * 64 + tj * 16][0], 40, lane);
      acc[0][tj] = wmma_bf16(a0, b, acc[0][tj]);
      acc[1][tj] = wmma_bf16(a1, b, acc[1][tj]);
    }
    __syncthreads();
  }

  const int nlo = lane & 15, mhi = (lane >> 4) << 3;
#pragma unroll
  for (int ti = 0; ti < 2; ++ti)
#pragma unroll
    for (int tj = 0; tj < 4; ++tj)
#pragma unroll
      for (int j = 0; j < 8; ++j) {
        int gm = bm + wr * 32 + ti * 16 + j + mhi;
        int gn = bn + wc * 64 + tj * 16 + nlo;
        float v = acc[ti][tj][j];
        if (mode == 0) {
          __builtin_nontemporal_store(v + bias[gn], &outF[(size_t)gm * ldb + gn]);
        } else {
          int b = gm >> 11, t = gm & 2047;
          int h = gn >> 6, d = gn & 63;
          if (mode == 1)
            outB[(((size_t)(b * 16 + h)) * 2048 + t) * 64 + d] = (bf16)(v * scale);
          else
            outB[(((size_t)(b * 16 + h)) * 64 + d) * 2048 + t] = (bf16)v;
        }
      }
}

// ---- flash attention: S^T = K x Q^T (WMMA), online softmax, O = P x V ----
// grid (qb=32, bh=32), 128 thr (4 waves); wave owns 16 q-rows, block 64.
__global__ void __launch_bounds__(128)
attn_flash(const bf16* __restrict__ qh, const bf16* __restrict__ kh,
           const bf16* __restrict__ vt, bf16* __restrict__ outb) {
  __shared__ bf16 sK[32][72];  // [key][d]
  __shared__ bf16 sV[64][40];  // [d][key] (V pre-transposed)
  const int tid = threadIdx.x, lane = tid & 31, wave = tid >> 5;
  const int qb = blockIdx.x, bh = blockIdx.y;
  const int qc = qb * 64 + wave * 16;
  const int nlo = lane & 15, hi = lane >> 4;

  v16bf qf[2];  // Q as B-matrix of S^T (q pre-scaled by 1/8)
  {
    const bf16* qrow = qh + ((size_t)bh * 2048 + qc + nlo) * 64 + (hi << 3);
    Frag f0, f1;
#pragma unroll
    for (int i = 0; i < 8; ++i) {
      f0.u[i] = *(const unsigned*)(qrow + ((i & 4) << 2) + ((i & 3) << 1));
      f1.u[i] = *(const unsigned*)(qrow + 32 + ((i & 4) << 2) + ((i & 3) << 1));
    }
    qf[0] = f0.v; qf[1] = f1.v;
  }

  float mrun = -INFINITY, lrun = 0.f;
  v8f o[4] = {};
  const int kr = tid >> 2, ksg = (tid & 3) * 16;
  const int vr = tid >> 1, vsg = (tid & 1) * 16;
  const int nkb = 2 * qb + 2;

  for (int kb = 0; kb < nkb; ++kb) {
    const int keyb = kb * 32;
    const bf16* gk = kh + ((size_t)bh * 2048 + keyb + kr) * 64 + ksg;
    async_copy16(&sK[kr][ksg], gk);
    async_copy16(&sK[kr][ksg + 8], gk + 8);
    const bf16* gv = vt + ((size_t)bh * 64 + vr) * 2048 + keyb + vsg;
    async_copy16(&sV[vr][vsg], gv);
    async_copy16(&sV[vr][vsg + 8], gv + 8);
    async_wait0();
    __syncthreads();

    v16bf ka0 = load_a_frag(&sK[0][0],   72, lane);
    v16bf ka1 = load_a_frag(&sK[0][32],  72, lane);
    v16bf kb0 = load_a_frag(&sK[16][0],  72, lane);
    v16bf kb1 = load_a_frag(&sK[16][32], 72, lane);
    v8f sa = {}, sb = {};
    sa = wmma_bf16(ka0, qf[0], sa); sa = wmma_bf16(ka1, qf[1], sa);
    sb = wmma_bf16(kb0, qf[0], sb); sb = wmma_bf16(kb1, qf[1], sb);

    const int qg = qc + nlo;  // C layout of S^T: M=key, N=q(lane)
#pragma unroll
    for (int j = 0; j < 8; ++j) {
      int kg = keyb + j + (hi << 3);
      if (kg > qg)      sa[j] = -INFINITY;
      if (kg + 16 > qg) sb[j] = -INFINITY;
    }
    float ml = -INFINITY;
#pragma unroll
    for (int j = 0; j < 8; ++j) { ml = fmaxf(ml, sa[j]); ml = fmaxf(ml, sb[j]); }
    ml = fmaxf(ml, __shfl_xor(ml, 16));
    float mnew = fmaxf(mrun, ml);
    float rescale = __expf(mrun - mnew);
    float ps = 0.f;
#pragma unroll
    for (int j = 0; j < 8; ++j) {
      sa[j] = __expf(sa[j] - mnew); ps += sa[j];
      sb[j] = __expf(sb[j] - mnew); ps += sb[j];
    }
    ps += __shfl_xor(ps, 16);
    lrun = lrun * rescale + ps;
    mrun = mnew;

    float rs[8];  // broadcast per-q rescale to O rows (M=q in O layout)
#pragma unroll
    for (int j = 0; j < 8; ++j) rs[j] = __shfl(rescale, j + (hi << 3));
#pragma unroll
    for (int nt = 0; nt < 4; ++nt)
#pragma unroll
      for (int j = 0; j < 8; ++j) o[nt][j] *= rs[j];

    Frag pf;  // lane-local S^T(C-layout) -> P(A-layout) repack
#pragma unroll
    for (int j = 0; j < 8; ++j) { pf.h[j] = (bf16)sa[j]; pf.h[8 + j] = (bf16)sb[j]; }
#pragma unroll
    for (int nt = 0; nt < 4; ++nt) {
      v16bf vf = load_b_frag(&sV[nt * 16][0], 40, lane);
      o[nt] = wmma_bf16(pf.v, vf, o[nt]);
    }
    __syncthreads();
  }

  float inv = 1.f / lrun;
  float nv[8];
#pragma unroll
  for (int j = 0; j < 8; ++j) nv[j] = __shfl(inv, j + (hi << 3));
  const int b = bh >> 4, h = bh & 15;
#pragma unroll
  for (int nt = 0; nt < 4; ++nt)
#pragma unroll
    for (int j = 0; j < 8; ++j) {
      int t = qc + j + (hi << 3);
      outb[((size_t)(b * 2048 + t)) * 1024 + h * 64 + nt * 16 + nlo] =
          (bf16)(o[nt][j] * nv[j]);
    }
}

// ---- per-row online log-sum-exp + NLL ------------------------------------
__global__ void __launch_bounds__(256)
loss_kernel(const float* __restrict__ logits, const int* __restrict__ target,
            float* __restrict__ loss) {
  __shared__ float sm[256], ss[256];
  const int row = blockIdx.x, tid = threadIdx.x;
  const float* lr = logits + (size_t)row * 32000;
  float m = -INFINITY, s = 0.f;
  for (int c = tid; c < 32000; c += 256) {
    float xv = lr[c];
    if (xv > m) { s = s * __expf(m - xv) + 1.f; m = xv; }
    else        { s += __expf(xv - m); }
  }
  sm[tid] = m; ss[tid] = s;
  __syncthreads();
  for (int off = 128; off > 0; off >>= 1) {
    if (tid < off) {
      float m2 = sm[tid + off], s2 = ss[tid + off];
      float M = fmaxf(sm[tid], m2);
      ss[tid] = ss[tid] * __expf(sm[tid] - M) + s2 * __expf(m2 - M);
      sm[tid] = M;
    }
    __syncthreads();
  }
  if (tid == 0) {
    float lse = __logf(ss[0]) + sm[0];
    atomicAdd(loss, (lse - lr[target[row]]) * (1.f / 4096.f));
  }
}

extern "C" void kernel_launch(void* const* d_in, const int* in_sizes, int n_in,
                              void* d_out, int out_size, void* d_ws, size_t ws_size,
                              hipStream_t stream) {
  (void)in_sizes; (void)n_in; (void)out_size; (void)ws_size;
  const int*   x      = (const int*)d_in[0];
  const int*   target = (const int*)d_in[1];
  const float* tok    = (const float*)d_in[2];
  const float* pos    = (const float*)d_in[3];
  const float* Wq     = (const float*)d_in[4];
  const float* Wk     = (const float*)d_in[5];
  const float* Wv     = (const float*)d_in[6];
  const float* Wlm    = (const float*)d_in[7];
  const float* blm    = (const float*)d_in[8];
  float* logits = (float*)d_out;
  float* loss   = logits + (size_t)4096 * 32000;

  char* ws = (char*)d_ws;
  size_t off = 0;
  auto alloc = [&](size_t bytes) {
    void* p = ws + off; off += (bytes + 255) & ~(size_t)255; return p;
  };
  bf16* h_bf   = (bf16*)alloc((size_t)4096 * 1024 * 2);
  bf16* wq_bf  = (bf16*)alloc((size_t)1024 * 1024 * 2);
  bf16* wk_bf  = (bf16*)alloc((size_t)1024 * 1024 * 2);
  bf16* wv_bf  = (bf16*)alloc((size_t)1024 * 1024 * 2);
  bf16* wlm_bf = (bf16*)alloc((size_t)1024 * 32000 * 2);
  bf16* q_h    = (bf16*)alloc((size_t)32 * 2048 * 64 * 2);
  bf16* k_h    = (bf16*)alloc((size_t)32 * 2048 * 64 * 2);
  bf16* v_t    = (bf16*)alloc((size_t)32 * 64 * 2048 * 2);
  bf16* a_out  = (bf16*)alloc((size_t)4096 * 1024 * 2);

  zero_loss<<<1, 1, 0, stream>>>(loss);
  embed_bf16<<<16384, 256, 0, stream>>>(x, tok, pos, h_bf);
  cvt_bf16<<<(1024 * 1024 + 255) / 256, 256, 0, stream>>>(Wq, wq_bf, 1024L * 1024);
  cvt_bf16<<<(1024 * 1024 + 255) / 256, 256, 0, stream>>>(Wk, wk_bf, 1024L * 1024);
  cvt_bf16<<<(1024 * 1024 + 255) / 256, 256, 0, stream>>>(Wv, wv_bf, 1024L * 1024);
  cvt_bf16<<<(int)((1024L * 32000 + 255) / 256), 256, 0, stream>>>(Wlm, wlm_bf, 1024L * 32000);

  dim3 g1(1024 / 128, 4096 / 128);
  gemm_bf16_wmma<<<g1, 256, 0, stream>>>(h_bf, wq_bf, 1024, 1024, nullptr, q_h, nullptr, 1, 0.125f);
  gemm_bf16_wmma<<<g1, 256, 0, stream>>>(h_bf, wk_bf, 1024, 1024, nullptr, k_h, nullptr, 1, 1.0f);
  gemm_bf16_wmma<<<g1, 256, 0, stream>>>(h_bf, wv_bf, 1024, 1024, nullptr, v_t, nullptr, 2, 1.0f);

  dim3 ga(32, 32);
  attn_flash<<<ga, 128, 0, stream>>>(q_h, k_h, v_t, a_out);

  dim3 g2(32000 / 128, 4096 / 128);
  gemm_bf16_wmma<<<g2, 256, 0, stream>>>(a_out, wlm_bf, 1024, 32000, logits, nullptr, blm, 0, 1.0f);

  loss_kernel<<<4096, 256, 0, stream>>>(logits, target, loss);
}
